// MultiheadAttentionWeights_80891414053488
// MI455X (gfx1250) — matrix operand order repaired and checked by reading, not compiled, not run on hardware
//
#include <hip/hip_runtime.h>
#include <hip/hip_bf16.h>

// Fused gather + in-projection GEMM + blocked QK^T + bias + softmax for MI455X
// (gfx1250, wave32, WMMA). One workgroup per (batch, 8x8 window); the 201MB
// projection intermediate lives only in LDS. Matrix math via
// v_wmma_f32_16x16x32_f16 with manually-built fragments in contiguous k order
// (k = 16*half + s in both A and B). Softmax is done in-register with half-wave __shfl_xor
// reductions keyed off the known C/D fragment layout.

typedef __attribute__((ext_vector_type(16))) _Float16 v16h;
typedef __attribute__((ext_vector_type(8)))  _Float16 v8h;
typedef __attribute__((ext_vector_type(4)))  _Float16 v4h;
typedef __attribute__((ext_vector_type(8)))  float    v8f;
typedef __attribute__((ext_vector_type(4)))  float    v4f;

#define EMBED   256
#define NHEADS  8
#define QHD     24
#define INPROJ  384
#define QDIM    192
#define TOK     64     // tokens per 8x8 window
#define PSLOT   512    // padded proj channels: 8 heads * 32 (q) + 8 heads * 32 (k)
#define A_PITCH 264    // halves; 256 + 8 pad (bank-conflict rotation)
#define P_PITCH 520    // halves; 512 + 8 pad

union V16 { v16h v; v8h h[2]; };

// ---- prep: w_in (256x384 f32) -> w^T (384x256 f16) ----
__global__ void prep_wt(const float* __restrict__ w_in, _Float16* __restrict__ wt) {
    int i = blockIdx.x * 256 + threadIdx.x;        // [0, 384*256)
    int n = i >> 8, k = i & 255;
    wt[i] = (_Float16)w_in[k * INPROJ + n];
}

// ---- prep: bias_g[h][m][n] = table[rpi[m*64+n]][h] ----
__global__ void prep_bias(const float* __restrict__ table, const int* __restrict__ rpi,
                          float* __restrict__ bias) {
    int i = blockIdx.x * 256 + threadIdx.x;        // [0, 8*4096)
    int h = i >> 12, mn = i & 4095;
    bias[i] = table[rpi[mn] * NHEADS + h];
}

__global__ __launch_bounds__(256)
void attn_fused(const float* __restrict__ x,
                const long long* __restrict__ idx,
                const float* __restrict__ b_in,
                const _Float16* __restrict__ wt,     // [384][256] f16 (w_in^T)
                const float* __restrict__ biasg,     // [8][64][64] f32
                float* __restrict__ out) {
    __shared__ _Float16 Ax[TOK * A_PITCH];   // gathered f16 inputs, token-major
    __shared__ _Float16 P [TOK * P_PITCH];   // projection, token-major, head-padded

    const int t    = threadIdx.x;
    const int lane = t & 31;
    const int wave = t >> 5;
    const int bb   = blockIdx.x;             // b*1024 + blk
    const int b    = bb >> 10;
    const int blk  = bb & 1023;
    const int bh   = blk >> 5, bw = blk & 31;
    const int lhi  = lane >> 4;              // 0/1 half-wave
    const int llo  = lane & 15;

    // ---- zero P (pad channels 24..31 of each head slot must be 0) ----
    {
        uint32_t* p32 = (uint32_t*)P;
        const int n32 = TOK * P_PITCH / 2;
        for (int i = t; i < n32; i += 256) p32[i] = 0u;
    }

    // ---- gather 64 token rows, f32 -> f16, into LDS (4 threads / token) ----
    {
        int tok = t >> 2;
        int i = tok >> 3, j = tok & 7;
        int chb = (t & 3) * 64;
        long long src = idx[(bh * 8 + i) * 256 + (bw * 8 + j)];
        const float* xr = x + ((size_t)b * 65536 + (size_t)src) * EMBED + chb;
        _Float16* ar = Ax + tok * A_PITCH + chb;
        #pragma unroll
        for (int u = 0; u < 16; ++u) {
            v4f f = ((const v4f*)xr)[u];
            v4h h4;
            h4[0] = (_Float16)f[0]; h4[1] = (_Float16)f[1];
            h4[2] = (_Float16)f[2]; h4[3] = (_Float16)f[3];
            ((v4h*)ar)[u] = h4;
        }
    }
    __syncthreads();

    // ================= GEMM1: P = Ax @ w_in + b_in =================
    // 24 N-tiles of 16; wave handles 3 of them, all 4 M-tiles, K=256.
    for (int q = 0; q < 3; ++q) {
        const int nt = wave * 3 + q;
        const int n  = nt * 16 + llo;                // this lane's output column
        const float bn = b_in[n];
        v8f acc[4];
        #pragma unroll
        for (int mt = 0; mt < 4; ++mt)
            #pragma unroll
            for (int v = 0; v < 8; ++v) acc[mt][v] = bn;

        const _Float16* wrow = wt + n * EMBED + lhi * 16;   // B frag: K contiguous
        for (int k0 = 0; k0 < 256; k0 += 32) {
            V16 bf;
            bf.h[0] = *(const v8h*)(wrow + k0);
            bf.h[1] = *(const v8h*)(wrow + k0 + 8);
            #pragma unroll
            for (int mt = 0; mt < 4; ++mt) {
                const _Float16* ap = Ax + (mt * 16 + llo) * A_PITCH + k0 + lhi * 16;
                V16 af;
                af.h[0] = *(const v8h*)ap;           // K = k0 + lhi*16 + [0..7]
                af.h[1] = *(const v8h*)(ap + 8);     // K = k0 + lhi*16 + 8 + [0..7]
                acc[mt] = __builtin_amdgcn_wmma_f32_16x16x32_f16(
                    false, af.v, false, bf.v, (short)0, acc[mt], false, false);
            }
        }
        // scatter column n into head-padded token-major P (f16)
        int slot;
        if (n < QDIM) slot = (n / QHD) * 32 + n % QHD;
        else { int n2 = n - QDIM; slot = 256 + (n2 / QHD) * 32 + n2 % QHD; }
        #pragma unroll
        for (int mt = 0; mt < 4; ++mt)
            #pragma unroll
            for (int v = 0; v < 8; ++v) {
                int m = mt * 16 + lhi * 8 + v;
                P[m * P_PITCH + slot] = (_Float16)acc[mt][v];
            }
    }
    __syncthreads();

    // ============ GEMM2 + softmax: wave == head, S = q @ k^T ============
    {
        const int h = wave;
        V16 bf[4];                                   // key fragments, 4 N-tiles
        #pragma unroll
        for (int tn = 0; tn < 4; ++tn) {
            const _Float16* kp = P + (tn * 16 + llo) * P_PITCH + 256 + h * 32 + lhi * 16;
            bf[tn].h[0] = *(const v8h*)kp;
            bf[tn].h[1] = *(const v8h*)(kp + 8);
        }
        const size_t obase   = (((size_t)h * 2 + b) * 1024 + (size_t)blk) * 4096;
        const float* bh_bias = biasg + h * 4096;

        for (int tm = 0; tm < 4; ++tm) {
            const _Float16* qp = P + (tm * 16 + llo) * P_PITCH + h * 32 + lhi * 16;
            V16 af;
            af.h[0] = *(const v8h*)qp;
            af.h[1] = *(const v8h*)(qp + 8);

            v8f c[4];
            #pragma unroll
            for (int tn = 0; tn < 4; ++tn) {
                v8f z = {};
                c[tn] = __builtin_amdgcn_wmma_f32_16x16x32_f16(
                    false, af.v, false, bf[tn].v, (short)0, z, false, false);
                #pragma unroll
                for (int v = 0; v < 8; ++v) {
                    int m = tm * 16 + lhi * 8 + v;
                    int n = tn * 16 + llo;
                    c[tn][v] += bh_bias[m * 64 + n];
                }
            }
            // rows live across the 16 lanes of each half-wave: reduce xor 1,2,4,8
            float mx[8], rs[8];
            #pragma unroll
            for (int v = 0; v < 8; ++v) {
                float m0 = fmaxf(fmaxf(c[0][v], c[1][v]), fmaxf(c[2][v], c[3][v]));
                m0 = fmaxf(m0, __shfl_xor(m0, 1));
                m0 = fmaxf(m0, __shfl_xor(m0, 2));
                m0 = fmaxf(m0, __shfl_xor(m0, 4));
                m0 = fmaxf(m0, __shfl_xor(m0, 8));
                mx[v] = m0;
            }
            #pragma unroll
            for (int v = 0; v < 8; ++v) {
                float s = 0.f;
                #pragma unroll
                for (int tn = 0; tn < 4; ++tn) {
                    float e = __expf(c[tn][v] - mx[v]);
                    c[tn][v] = e;
                    s += e;
                }
                s += __shfl_xor(s, 1);
                s += __shfl_xor(s, 2);
                s += __shfl_xor(s, 4);
                s += __shfl_xor(s, 8);
                rs[v] = 1.0f / s;
            }
            #pragma unroll
            for (int tn = 0; tn < 4; ++tn)
                #pragma unroll
                for (int v = 0; v < 8; ++v) {
                    int m = tm * 16 + lhi * 8 + v;
                    int n = tn * 16 + llo;
                    out[obase + (size_t)(m * 64 + n)] = c[tn][v] * rs[v];
                }
        }
    }
}

extern "C" void kernel_launch(void* const* d_in, const int* in_sizes, int n_in,
                              void* d_out, int out_size, void* d_ws, size_t ws_size,
                              hipStream_t stream) {
    const float*     x     = (const float*)d_in[0];
    const long long* idx   = (const long long*)d_in[1];   // int64 indices
    const float*     w_in  = (const float*)d_in[2];
    const float*     b_in  = (const float*)d_in[3];
    const float*     table = (const float*)d_in[4];
    const int*       rpi   = (const int*)d_in[5];

    // workspace: [0, 192KB) w^T f16 ; [192KB, 320KB) bias table f32
    _Float16* wt    = (_Float16*)d_ws;
    float*    biasg = (float*)((char*)d_ws + (size_t)INPROJ * EMBED * sizeof(_Float16));

    prep_wt  <<<(INPROJ * EMBED) / 256, 256, 0, stream>>>(w_in, wt);
    prep_bias<<<(NHEADS * 4096)  / 256, 256, 0, stream>>>(table, rpi, biasg);
    attn_fused<<<2 * 1024, 256, 0, stream>>>(x, idx, b_in, wt, biasg, (float*)d_out);
}
